// STransformer_5634997092989
// MI455X (gfx1250) — matrix operand order repaired and checked by reading, not compiled, not run on hardware
//
#include <hip/hip_runtime.h>

// ---------------------------------------------------------------------------
// STransformer forward for MI455X (gfx1250, wave32, WMMA).
// Tokens R = B*N*T = 163840, C = 256. All big GEMMs run on
// v_wmma_f32_16x16x32_bf16 (bf16 operands, f32 accumulate). Weights are
// packed once per call into the wave32 B-fragment layout so the inner loop is
// one coalesced 32B load per lane per K-step. Each wave computes a 16x64
// output strip (4 WMMAs per K-step). The K-loop is unrolled by 2 with two
// explicit register buffer sets (no staging copies): each half-iteration
// converts+WMMAs out of one set, then refills that set for kt+2 while the
// WMMAs execute. fp32->bf16 via v_perm_b32 packing (3 VALU per 2 elements).
// Bias/residual/ReLU are compile-time template flags -> branch-free epilogue.
// Attention over the N=5 agent axis + adj mixing + LN are VALU kernels.
// ---------------------------------------------------------------------------

constexpr int B_  = 64;
constexpr int N_  = 5;
constexpr int T_  = 512;
constexpr int C_  = 256;
constexpr int H_  = 4;
constexpr int DH_ = 64;
constexpr int R_  = B_ * N_ * T_;                 // 163840 token rows
constexpr long long RC_ = (long long)R_ * C_;     // 41,943,040 floats

typedef __attribute__((ext_vector_type(16))) __bf16   v16bf;
typedef __attribute__((ext_vector_type(8)))  float    v8f;
typedef __attribute__((ext_vector_type(8)))  unsigned v8u;
typedef __attribute__((ext_vector_type(4)))  float    v4f;

__device__ __forceinline__ __bf16 f2bf(float f) {      // RNE (pack kernel only)
  union { float f; unsigned u; } x; x.f = f;
  unsigned r = x.u + 0x7FFFu + ((x.u >> 16) & 1u);
  unsigned short h = (unsigned short)(r >> 16);
  return __builtin_bit_cast(__bf16, h);
}

// two fp32 -> packed bf16x2 dword: round-half-up + v_perm_b32
__device__ __forceinline__ unsigned pk2bf(float lo, float hi) {
  unsigned a = __builtin_bit_cast(unsigned, lo) + 0x8000u;
  unsigned b = __builtin_bit_cast(unsigned, hi) + 0x8000u;
  return __builtin_amdgcn_perm(b, a, 0x07060302u);
}

__device__ __forceinline__ v16bf cvtA(v4f a0, v4f a1, v4f a2, v4f a3) {
  v8u au;
  au[0] = pk2bf(a0.x, a0.y); au[1] = pk2bf(a0.z, a0.w);
  au[2] = pk2bf(a1.x, a1.y); au[3] = pk2bf(a1.z, a1.w);
  au[4] = pk2bf(a2.x, a2.y); au[5] = pk2bf(a2.z, a2.w);
  au[6] = pk2bf(a3.x, a3.y); au[7] = pk2bf(a3.z, a3.w);
  return __builtin_bit_cast(v16bf, au);
}

// ---------------------------------------------------------------------------
// Pack W[K,Nn] (fp32 row-major) into WMMA B fragments:
//   dst[((nt*ktc + kt)*32 + lane)*16 + e] = bf16( W[kt*32 + g*16 + e][nt*16 + (lane&15)] )
// (k order used here, same as cvtA's A: lanes 0-15 carry K=0..15, lanes 16-31 carry K=16..31,
//  VGPR j holds the (2j,2j+1) K pair -> element e <-> k = g*16 + e.)
// ---------------------------------------------------------------------------
__global__ void pack_w_kernel(const float* __restrict__ W, __bf16* __restrict__ Wp,
                              int K, int Nn) {
  int ktc = K >> 5;
  int total = (Nn >> 4) * ktc * 32;
  int idx = blockIdx.x * blockDim.x + threadIdx.x;
  if (idx >= total) return;
  int lane = idx & 31;
  int kt   = (idx >> 5) % ktc;
  int nt   = idx / (32 * ktc);
  int g    = lane >> 4;
  int ncol = nt * 16 + (lane & 15);
  int kbase = kt * 32 + g * 16;
  __bf16* dst = Wp + (size_t)idx * 16;
#pragma unroll
  for (int e = 0; e < 16; ++e)
    dst[e] = f2bf(W[(size_t)(kbase + e) * Nn + ncol]);
}

// ---------------------------------------------------------------------------
// Generic GEMM: out[M,Nn] = A[M,K] @ W (+bias) (+resid), optional ReLU.
// 1 wave = one 16x64 output strip (4 WMMA n-tiles); 8 waves / block.
// Requires K % 64 == 0 (ktc even), Nn % 64 == 0, M % 16 == 0.
// ---------------------------------------------------------------------------
template <bool HAS_BIAS, bool HAS_RESID, bool DO_RELU>
__global__ void __launch_bounds__(256)
gemm_bf16_wmma(const float* __restrict__ A, const __bf16* __restrict__ Wp,
               const float* __restrict__ bias, const float* __restrict__ resid,
               float* __restrict__ out, int M, int K, int Nn) {
  const int lane = threadIdx.x & 31;
  const int wid  = blockIdx.x * 8 + (threadIdx.x >> 5);
  const int ngc  = Nn >> 6;                 // 64-wide n-groups
  const int ktc  = K >> 5;                  // even by construction
  const int tile_m = wid / ngc;
  const int ng     = wid - tile_m * ngc;    // which 64-col group
  const int g    = lane >> 4;
  const int row  = tile_m * 16 + (lane & 15);

  const float*  arow = A + (size_t)row * K + g * 16;
  const __bf16* bptr = Wp + ((size_t)(ng * 4) * ktc) * 512 + (size_t)lane * 16;
  const size_t  bstr = (size_t)ktc * 512;   // between adjacent n-tiles

  // two explicit buffer sets: no staging copies
  v4f  a00, a01, a02, a03, a10, a11, a12, a13;
  v16bf b00, b01, b02, b03, b10, b11, b12, b13;

#define LOAD_A(kt, x0, x1, x2, x3)                                          \
  { const float* p = arow + (kt) * 32;                                      \
    x0 = *(const v4f*)(p);      x1 = *(const v4f*)(p + 4);                  \
    x2 = *(const v4f*)(p + 8);  x3 = *(const v4f*)(p + 12); }
#define LOAD_B(kt, y0, y1, y2, y3)                                          \
  { const __bf16* bk = bptr + (size_t)(kt) * 512;                           \
    y0 = *(const v16bf*)(bk);            y1 = *(const v16bf*)(bk + bstr);   \
    y2 = *(const v16bf*)(bk + 2 * bstr); y3 = *(const v16bf*)(bk + 3 * bstr); }

  LOAD_A(0, a00, a01, a02, a03)
  LOAD_B(0, b00, b01, b02, b03)
  LOAD_A(1, a10, a11, a12, a13)
  LOAD_B(1, b10, b11, b12, b13)

  v8f acc0 = {}, acc1 = {}, acc2 = {}, acc3 = {};
  for (int kt = 0; kt < ktc; kt += 2) {
    // ---- half-iteration 0: consume set0, refill set0 for kt+2 ----
    v16bf af0 = cvtA(a00, a01, a02, a03);          // waits on set0 loads
    acc0 = __builtin_amdgcn_wmma_f32_16x16x32_bf16(false, af0, false, b00, (short)0, acc0, false, false);
    acc1 = __builtin_amdgcn_wmma_f32_16x16x32_bf16(false, af0, false, b01, (short)0, acc1, false, false);
    acc2 = __builtin_amdgcn_wmma_f32_16x16x32_bf16(false, af0, false, b02, (short)0, acc2, false, false);
    acc3 = __builtin_amdgcn_wmma_f32_16x16x32_bf16(false, af0, false, b03, (short)0, acc3, false, false);
    const int n0 = (kt + 2 < ktc) ? kt + 2 : kt;   // branch-free tail clamp
    LOAD_A(n0, a00, a01, a02, a03)
    LOAD_B(n0, b00, b01, b02, b03)
    __builtin_prefetch(arow + (n0 + 2) * 32, 0, 1);

    // ---- half-iteration 1: consume set1, refill set1 for kt+3 ----
    v16bf af1 = cvtA(a10, a11, a12, a13);          // waits on set1 loads
    acc0 = __builtin_amdgcn_wmma_f32_16x16x32_bf16(false, af1, false, b10, (short)0, acc0, false, false);
    acc1 = __builtin_amdgcn_wmma_f32_16x16x32_bf16(false, af1, false, b11, (short)0, acc1, false, false);
    acc2 = __builtin_amdgcn_wmma_f32_16x16x32_bf16(false, af1, false, b12, (short)0, acc2, false, false);
    acc3 = __builtin_amdgcn_wmma_f32_16x16x32_bf16(false, af1, false, b13, (short)0, acc3, false, false);
    const int n1 = (kt + 3 < ktc) ? kt + 3 : kt + 1;
    LOAD_A(n1, a10, a11, a12, a13)
    LOAD_B(n1, b10, b11, b12, b13)
  }
#undef LOAD_A
#undef LOAD_B

  v8f accs[4] = {acc0, acc1, acc2, acc3};
#pragma unroll
  for (int j = 0; j < 4; ++j) {
    const int ncol = (ng * 4 + j) * 16 + (lane & 15);
    float bv = 0.0f;
    if constexpr (HAS_BIAS) bv = bias[ncol];
#pragma unroll
    for (int r = 0; r < 8; ++r) {
      const int m = tile_m * 16 + g * 8 + r;   // C/D layout: VGPR r -> M = r + g*8
      float v = accs[j][r] + bv;
      if constexpr (HAS_RESID) v += resid[(size_t)m * Nn + ncol];
      if constexpr (DO_RELU)   v = fmaxf(v, 0.0f);
      out[(size_t)m * Nn + ncol] = v;
    }
  }
}

// ---------------------------------------------------------------------------
// q2 = query + (adj @ We + be), output in [B,N,T,C] row order.
// ---------------------------------------------------------------------------
__global__ void embed_kernel(const float* __restrict__ query, const float* __restrict__ adj,
                             const float* __restrict__ We, const float* __restrict__ be,
                             float* __restrict__ q2) {
  int row = blockIdx.x;                 // (b,n,t)
  int c   = threadIdx.x;                // 0..255
  int t = row % T_;
  int n = (row / T_) % N_;
  int b = row / (T_ * N_);
  const float* a = adj + (((size_t)b * T_ + t) * N_ + n) * N_;
  float acc = be[c];
#pragma unroll
  for (int m = 0; m < N_; ++m) acc += a[m] * We[m * C_ + c];
  size_t i = (size_t)row * C_ + c;
  q2[i] = query[i] + acc;
}

// ---------------------------------------------------------------------------
// adj mixing for GCN: out[b,n,t,:] = act( sum_m adj[b,t,n,m] * P[b,m,t,:] + bias )
// ---------------------------------------------------------------------------
__global__ void mix_kernel(const float* __restrict__ adj, const float* __restrict__ P,
                           const float* __restrict__ bias, float* __restrict__ out,
                           int Cp, int do_relu) {
  int row = blockIdx.x;                 // (b,n,t)
  int t = row % T_;
  int n = (row / T_) % N_;
  int b = row / (T_ * N_);
  float a[N_];
#pragma unroll
  for (int m = 0; m < N_; ++m)
    a[m] = adj[(((size_t)b * T_ + t) * N_ + n) * N_ + m];
  for (int c = threadIdx.x; c < Cp; c += blockDim.x) {
    float acc = bias[c];
#pragma unroll
    for (int m = 0; m < N_; ++m)
      acc += a[m] * P[((size_t)(b * N_ + m) * T_ + t) * Cp + c];
    if (do_relu) acc = fmaxf(acc, 0.0f);
    out[(size_t)row * Cp + c] = acc;
  }
}

// ---------------------------------------------------------------------------
// Spatial attention over the N=5 agent axis. One thread per (b,h,t).
// ---------------------------------------------------------------------------
__global__ void attn_kernel(const float* __restrict__ Q, const float* __restrict__ Kk,
                            const float* __restrict__ V, float* __restrict__ ctx) {
  int idx = blockIdx.x * blockDim.x + threadIdx.x;
  if (idx >= B_ * H_ * T_) return;
  int t = idx % T_;
  int h = (idx / T_) % H_;
  int b = idx / (T_ * H_);
  size_t rbase[N_];
#pragma unroll
  for (int n = 0; n < N_; ++n)
    rbase[n] = ((size_t)(b * N_ + n) * T_ + t) * C_ + h * DH_;

  float sc[N_][N_];
#pragma unroll
  for (int n = 0; n < N_; ++n) {
    float qv[DH_];
#pragma unroll
    for (int d = 0; d < DH_; ++d) qv[d] = Q[rbase[n] + d];
#pragma unroll
    for (int m = 0; m < N_; ++m) {
      float s = 0.0f;
#pragma unroll
      for (int d = 0; d < DH_; ++d) s += qv[d] * Kk[rbase[m] + d];
      sc[n][m] = s * 0.125f;             // 1/sqrt(64)
    }
  }
#pragma unroll
  for (int n = 0; n < N_; ++n) {
    float mx = sc[n][0];
#pragma unroll
    for (int m = 1; m < N_; ++m) mx = fmaxf(mx, sc[n][m]);
    float ss = 0.0f;
#pragma unroll
    for (int m = 0; m < N_; ++m) { sc[n][m] = __expf(sc[n][m] - mx); ss += sc[n][m]; }
    float inv = 1.0f / ss;
#pragma unroll
    for (int m = 0; m < N_; ++m) sc[n][m] *= inv;
  }
#pragma unroll 4
  for (int d = 0; d < DH_; ++d) {
    float vv[N_];
#pragma unroll
    for (int m = 0; m < N_; ++m) vv[m] = V[rbase[m] + d];
#pragma unroll
    for (int n = 0; n < N_; ++n) {
      float a = 0.0f;
#pragma unroll
      for (int m = 0; m < N_; ++m) a += sc[n][m] * vv[m];
      ctx[rbase[n] + d] = a;
    }
  }
}

// ---------------------------------------------------------------------------
// LayerNorm over C=256 (input already includes residual). 1 block = 1 row.
// ---------------------------------------------------------------------------
__global__ void ln_kernel(const float* __restrict__ in, const float* __restrict__ gam,
                          const float* __restrict__ bet, float* __restrict__ out) {
  __shared__ float red[256];
  int row = blockIdx.x, c = threadIdx.x;
  float v = in[(size_t)row * C_ + c];
  red[c] = v; __syncthreads();
  for (int s = 128; s > 0; s >>= 1) { if (c < s) red[c] += red[c + s]; __syncthreads(); }
  float mean = red[0] * (1.0f / C_); __syncthreads();
  float d = v - mean;
  red[c] = d * d; __syncthreads();
  for (int s = 128; s > 0; s >>= 1) { if (c < s) red[c] += red[c + s]; __syncthreads(); }
  float var = red[0] * (1.0f / C_);
  out[(size_t)row * C_ + c] = d * rsqrtf(var + 1e-5f) * gam[c] + bet[c];
}

// ---------------------------------------------------------------------------
// out = sigmoid(G1+G2) * U + (1 - sigmoid(G1+G2)) * XG
// ---------------------------------------------------------------------------
__global__ void gate_kernel(const float* __restrict__ G1, const float* __restrict__ G2,
                            const float* __restrict__ U, const float* __restrict__ XG,
                            float* __restrict__ out, long long total) {
  long long i = (long long)blockIdx.x * blockDim.x + threadIdx.x;
  if (i >= total) return;
  float g = 1.0f / (1.0f + __expf(-(G1[i] + G2[i])));
  out[i] = g * U[i] + (1.0f - g) * XG[i];
}

// ---------------------------------------------------------------------------
// Host launch
// ---------------------------------------------------------------------------
static inline void launch_gemm(const float* A, const __bf16* Wp, const float* bias,
                               const float* resid, float* out, int M, int K, int Nn,
                               int relu, hipStream_t s) {
  int blocks = (M / 16) * (Nn / 64) / 8;
  if (bias && resid)
    gemm_bf16_wmma<true, true, false><<<blocks, 256, 0, s>>>(A, Wp, bias, resid, out, M, K, Nn);
  else if (bias && relu)
    gemm_bf16_wmma<true, false, true><<<blocks, 256, 0, s>>>(A, Wp, bias, resid, out, M, K, Nn);
  else if (bias)
    gemm_bf16_wmma<true, false, false><<<blocks, 256, 0, s>>>(A, Wp, bias, resid, out, M, K, Nn);
  else
    gemm_bf16_wmma<false, false, false><<<blocks, 256, 0, s>>>(A, Wp, bias, resid, out, M, K, Nn);
}

static inline void launch_pack(const float* W, __bf16* Wp, int K, int Nn, hipStream_t s) {
  int total = (Nn / 16) * (K / 32) * 32;
  pack_w_kernel<<<(total + 255) / 256, 256, 0, s>>>(W, Wp, K, Nn);
}

extern "C" void kernel_launch(void* const* d_in, const int* in_sizes, int n_in,
                              void* d_out, int out_size, void* d_ws, size_t ws_size,
                              hipStream_t stream) {
  (void)in_sizes; (void)n_in; (void)out_size; (void)ws_size;
  // setup_inputs() order:
  const float* query = (const float*)d_in[2];   // value(0), key(1) are unused by ref
  const float* adj   = (const float*)d_in[3];
  const float* Wq  = (const float*)d_in[4];
  const float* Wk  = (const float*)d_in[5];
  const float* Wv  = (const float*)d_in[6];
  const float* Wo  = (const float*)d_in[7];
  const float* bo  = (const float*)d_in[8];
  const float* We  = (const float*)d_in[9];
  const float* be  = (const float*)d_in[10];
  const float* g1  = (const float*)d_in[11];
  const float* b1  = (const float*)d_in[12];
  const float* g2  = (const float*)d_in[13];
  const float* b2  = (const float*)d_in[14];
  const float* Wf1 = (const float*)d_in[15];
  const float* bf1 = (const float*)d_in[16];
  const float* Wf2 = (const float*)d_in[17];
  const float* bf2 = (const float*)d_in[18];
  const float* Wg1 = (const float*)d_in[19];
  const float* bg1 = (const float*)d_in[20];
  const float* Wg2 = (const float*)d_in[21];
  const float* bg2 = (const float*)d_in[22];
  const float* Wfs = (const float*)d_in[23];
  const float* bfs = (const float*)d_in[24];
  const float* Wfg = (const float*)d_in[25];
  const float* bfg = (const float*)d_in[26];

  float* ws = (float*)d_ws;
  float* q2 = ws;                     // R x C
  float* XG = ws + RC_;               // R x C
  float* U  = ws + 2 * RC_;           // R x C
  float* T1 = ws + 3 * RC_;           // R x 4C scratch
  float* T2 = ws + 7 * RC_;           // R x 4C scratch

  __bf16* PK   = (__bf16*)(ws + 11 * RC_);
  __bf16* Wg1p = PK;                  // 256x512
  __bf16* Wg2p = Wg1p + 131072;       // 512x256
  __bf16* Wqp  = Wg2p + 131072;
  __bf16* Wkp  = Wqp  + 65536;
  __bf16* Wvp  = Wkp  + 65536;
  __bf16* Wop  = Wvp  + 65536;
  __bf16* Wf1p = Wop  + 65536;        // 256x1024
  __bf16* Wf2p = Wf1p + 262144;       // 1024x256
  __bf16* Wfsp = Wf2p + 262144;
  __bf16* Wfgp = Wfsp + 65536;

  // pack all weight matrices into WMMA-B layout (cheap, ~2.4 MB total)
  launch_pack(Wg1, Wg1p, C_, 2 * C_, stream);
  launch_pack(Wg2, Wg2p, 2 * C_, C_, stream);
  launch_pack(Wq,  Wqp,  C_, C_, stream);
  launch_pack(Wk,  Wkp,  C_, C_, stream);
  launch_pack(Wv,  Wvp,  C_, C_, stream);
  launch_pack(Wo,  Wop,  C_, C_, stream);
  launch_pack(Wf1, Wf1p, C_, 4 * C_, stream);
  launch_pack(Wf2, Wf2p, 4 * C_, C_, stream);
  launch_pack(Wfs, Wfsp, C_, C_, stream);
  launch_pack(Wfg, Wfgp, C_, C_, stream);

  // q2 = query + spatial embedding of adjacency
  embed_kernel<<<R_, 256, 0, stream>>>(query, adj, We, be, q2);

  // GCN: h = relu(adj @ (query @ Wg1) + bg1); X_G = adj @ (h @ Wg2) + bg2
  launch_gemm(query, Wg1p, nullptr, nullptr, T1, R_, C_, 2 * C_, 0, stream);
  mix_kernel<<<R_, 256, 0, stream>>>(adj, T1, bg1, T2, 2 * C_, 1);          // h in T2
  launch_gemm(T2, Wg2p, nullptr, nullptr, T1, R_, 2 * C_, C_, 0, stream);
  mix_kernel<<<R_, 256, 0, stream>>>(adj, T1, bg2, XG, C_, 0);

  // Q, K, V projections of q2
  float* Qb = T1;
  float* Kb = T1 + RC_;
  float* Vb = T1 + 2 * RC_;
  float* Cx = T1 + 3 * RC_;
  launch_gemm(q2, Wqp, nullptr, nullptr, Qb, R_, C_, C_, 0, stream);
  launch_gemm(q2, Wkp, nullptr, nullptr, Kb, R_, C_, C_, 0, stream);
  launch_gemm(q2, Wvp, nullptr, nullptr, Vb, R_, C_, C_, 0, stream);

  // softmax attention over N=5 agents
  attn_kernel<<<(B_ * H_ * T_ + 255) / 256, 256, 0, stream>>>(Qb, Kb, Vb, Cx);

  // attention out projection + residual, LN1
  launch_gemm(Cx, Wop, bo, q2, T2, R_, C_, C_, 0, stream);
  float* xb = T2 + RC_;
  ln_kernel<<<R_, 256, 0, stream>>>(T2, g1, b1, xb);

  // feed-forward: relu(x@Wf1+bf1)@Wf2+bf2 + x, LN2 -> U_S
  launch_gemm(xb, Wf1p, bf1, nullptr, T1, R_, C_, 4 * C_, 1, stream);       // ff1 (R x 1024)
  launch_gemm(T1, Wf2p, bf2, xb, T2 + 2 * RC_, R_, 4 * C_, C_, 0, stream);
  ln_kernel<<<R_, 256, 0, stream>>>(T2 + 2 * RC_, g2, b2, U);

  // gate: sigmoid(U@Wfs + bfs + XG@Wfg + bfg)
  launch_gemm(U,  Wfsp, bfs, nullptr, T2,       R_, C_, C_, 0, stream);     // G1
  launch_gemm(XG, Wfgp, bfg, nullptr, T2 + RC_, R_, C_, C_, 0, stream);     // G2
  gate_kernel<<<(int)(RC_ / 256), 256, 0, stream>>>(T2, T2 + RC_, U, XG,
                                                    (float*)d_out, RC_);
}
